// LatentDynamicsModel_79568564126438
// MI455X (gfx1250) — matrix operand order repaired and checked by reading, not compiled, not run on hardware
//
#include <hip/hip_runtime.h>
#include <hip/hip_bf16.h>

// ---------------- problem constants ----------------
#define BB   512
#define TT   32
#define OBS  512
#define ACT  32
#define LAT  1024
#define HID  2048
#define RHID 1024
#define RIN  1056          // LAT + ACT
#define GN   6144          // 3*LAT (gi) + 3*LAT (gh) fused

typedef __attribute__((ext_vector_type(16))) __bf16 v16bf;
typedef __attribute__((ext_vector_type(8)))  float  v8f;

// fp32 -> bf16 (round-to-nearest-even), pure bit math
__device__ __forceinline__ unsigned f2bf(float f) {
    unsigned u = __builtin_bit_cast(unsigned, f);
    return (u + 0x7FFFu + ((u >> 16) & 1u)) >> 16;
}
__device__ __forceinline__ unsigned pack2(float lo, float hi) {
    return f2bf(lo) | (f2bf(hi) << 16);
}

// ---------------- packing kernels ----------------
// A-fragment packing (ISA 7.12.2, 16-bit A 16x32):
// tile (mt,kt): lane L (m=L&15, half=L>>4), vgpr v: k = kt*32 + (v/4)*16 + half*8 + (v%4)*2
__global__ __launch_bounds__(256) void pack_a_k(const float* __restrict__ src,
                                                unsigned* __restrict__ dst, int K) {
    int idx  = blockIdx.x * 256 + threadIdx.x;      // one packed dword (2 bf16)
    int KT   = K >> 5;
    int tile = idx >> 8;
    int r    = idx & 255;
    int lane = r >> 3, v = r & 7;
    int mt = tile / KT, kt = tile % KT;
    int m    = mt * 16 + (lane & 15);
    int half = lane >> 4;
    int k    = kt * 32 + (v >> 2) * 16 + half * 8 + (v & 3) * 2;
    const float* s = src + m * K + k;
    dst[idx] = pack2(s[0], s[1]);
}

// fused concat + A-pack for x = [h | a_t]  (K = RIN = 1056, KT = 33)
__global__ __launch_bounds__(256) void pack_ax_k(const float* __restrict__ h,
                                                 const float* __restrict__ act,
                                                 int t, unsigned* __restrict__ dst) {
    const int KT = RIN >> 5;
    int idx  = blockIdx.x * 256 + threadIdx.x;
    int tile = idx >> 8;
    int r    = idx & 255;
    int lane = r >> 3, v = r & 7;
    int mt = tile / KT, kt = tile % KT;
    int m    = mt * 16 + (lane & 15);
    int half = lane >> 4;
    int k    = kt * 32 + (v >> 2) * 16 + half * 8 + (v & 3) * 2;
    float f0, f1;
    if (k < LAT) {                       // pair never straddles LAT (even boundary)
        const float* s = h + m * LAT + k;
        f0 = s[0]; f1 = s[1];
    } else {
        const float* s = act + m * TT * ACT + t * ACT + (k - LAT);
        f0 = s[0]; f1 = s[1];
    }
    dst[idx] = pack2(f0, f1);
}

// B-fragment packing (32x16 K-tile): lane L (n=L&15, half=L>>4), vgpr j: k = kt*32 + (j/4)*16 + half*8 + (j%4)*2
__global__ __launch_bounds__(256) void pack_b_k(const float* __restrict__ src,
                                                unsigned* __restrict__ dst, int K, int N) {
    int idx  = blockIdx.x * 256 + threadIdx.x;
    int KT   = K >> 5;
    int tile = idx >> 8;
    int r    = idx & 255;
    int lane = r >> 3, j = r & 7;
    int nt = tile / KT, kt = tile % KT;
    int n = nt * 16 + (lane & 15);
    int k = kt * 32 + (j >> 2) * 16 + (lane >> 4) * 8 + (j & 3) * 2;
    dst[idx] = pack2(src[k * N + n], src[(k + 1) * N + n]);
}

// fused GRU weight: cols [0,3072) = wih (K=1056), cols [3072,6144) = whh (K=1024, pad 0)
__global__ __launch_bounds__(256) void pack_b_gru_k(const float* __restrict__ wih,
                                                    const float* __restrict__ whh,
                                                    unsigned* __restrict__ dst) {
    const int KT = RIN >> 5;
    int idx  = blockIdx.x * 256 + threadIdx.x;
    int tile = idx >> 8;
    int r    = idx & 255;
    int lane = r >> 3, j = r & 7;
    int nt = tile / KT, kt = tile % KT;
    int n = nt * 16 + (lane & 15);
    int k = kt * 32 + (j >> 2) * 16 + (lane >> 4) * 8 + (j & 3) * 2;
    float f0, f1;
    if (n < 3 * LAT) {
        f0 = wih[k * (3 * LAT) + n];
        f1 = wih[(k + 1) * (3 * LAT) + n];
    } else {
        int nn = n - 3 * LAT;
        f0 = (k     < LAT) ? whh[k       * (3 * LAT) + nn] : 0.f;
        f1 = (k + 1 < LAT) ? whh[(k + 1) * (3 * LAT) + nn] : 0.f;
    }
    dst[idx] = pack2(f0, f1);
}

__global__ __launch_bounds__(256) void gbias_k(const float* __restrict__ bih,
                                               const float* __restrict__ bhh,
                                               float* __restrict__ gb) {
    int i = blockIdx.x * 256 + threadIdx.x;
    gb[i] = (i < 3 * LAT) ? bih[i] : bhh[i - 3 * LAT];
}

// ---------------- WMMA GEMM: C[M=512,N] = relu?(Apack @ Bpack + bias) ----------------
// block = 128 threads (4 waves, 2x2), block tile 128x128, wave tile 64x64 (4x4 WMMA tiles)
// KT is compile-time so fragment-row strides fold into immediate offsets; per-iter
// address math reduces to two pointer increments of 1024 B (one K-tile).
template <int KT>
__global__ __launch_bounds__(128) void gemm_bf16_wmma(const char* __restrict__ Ap,
                                                      const char* __restrict__ Bp,
                                                      const float* __restrict__ bias,
                                                      float* __restrict__ C,
                                                      int ldc, int relu) {
    const int lane = threadIdx.x & 31;
    const int wv   = threadIdx.x >> 5;
    const int wr   = wv >> 1, wc = wv & 1;
    const int mt0  = blockIdx.y * 8 + wr * 4;   // in 16-row tiles
    const int nt0  = blockIdx.x * 8 + wc * 4;   // in 16-col tiles

    const char* aB = Ap + (size_t)mt0 * (KT * 1024) + lane * 32;
    const char* bB = Bp + (size_t)nt0 * (KT * 1024) + lane * 32;

    v8f acc[4][4];
#pragma unroll
    for (int i = 0; i < 4; ++i)
#pragma unroll
        for (int j = 0; j < 4; ++j)
            acc[i][j] = (v8f){0.f, 0.f, 0.f, 0.f, 0.f, 0.f, 0.f, 0.f};

#pragma unroll 1
    for (int kt = 0; kt < KT; ++kt) {
        v16bf a[4], b[4];
#pragma unroll
        for (int i = 0; i < 4; ++i)
            a[i] = *(const v16bf*)(aB + i * (KT * 1024));
#pragma unroll
        for (int j = 0; j < 4; ++j)
            b[j] = *(const v16bf*)(bB + j * (KT * 1024));
        if (kt + 1 < KT) {   // uniform: warm next K-tile (global_prefetch)
            __builtin_prefetch(aB + 1024, 0, 1);
            __builtin_prefetch(bB + 1024, 0, 1);
        }
#pragma unroll
        for (int i = 0; i < 4; ++i)
#pragma unroll
            for (int j = 0; j < 4; ++j)
                acc[i][j] = __builtin_amdgcn_wmma_f32_16x16x32_bf16(
                    false, a[i], false, b[j], (short)0, acc[i][j], false, false);
        aB += 1024;
        bB += 1024;
    }

    // epilogue: C/D layout — vgpr c: row = tile*16 + (lane>>4)*8 + c, col = tile*16 + (lane&15)
    const int half = lane >> 4, nl = lane & 15;
#pragma unroll
    for (int i = 0; i < 4; ++i) {
#pragma unroll
        for (int j = 0; j < 4; ++j) {
            int col = (nt0 + j) * 16 + nl;
            float bv = bias[col];
#pragma unroll
            for (int c = 0; c < 8; ++c) {
                int row = (mt0 + i) * 16 + half * 8 + c;
                float v = acc[i][j][c] + bv;
                v = relu ? fmaxf(v, 0.f) : v;
                C[row * ldc + col] = v;
            }
        }
    }
}

// ---------------- GRU gate fusion: h = (1-z)*n + z*h, also emit bf16 A-pack ----------------
// one thread per pair (j, j+1); writes fp32 h (float2) and one packed dword of hpack
__global__ __launch_bounds__(256) void gru_gate_k(const float* __restrict__ g,
                                                  float* __restrict__ h,
                                                  unsigned* __restrict__ hpack) {
    int idx = blockIdx.x * 256 + threadIdx.x;       // over B*LAT/2
    int b = idx >> 9, jp = idx & 511;
    int j = jp * 2;
    const float* gb = g + b * GN;
    float hn2[2];
#pragma unroll
    for (int e = 0; e < 2; ++e) {
        int jj = j + e;
        float ir = gb[jj], iz = gb[LAT + jj], in = gb[2 * LAT + jj];
        float hr = gb[3 * LAT + jj], hz = gb[4 * LAT + jj], hnv = gb[5 * LAT + jj];
        float r = 1.f / (1.f + __expf(-(ir + hr)));
        float z = 1.f / (1.f + __expf(-(iz + hz)));
        float n = tanhf(in + r * hnv);
        float hp = h[b * LAT + jj];
        hn2[e] = (1.f - z) * n + z * hp;
    }
    *(float2*)(h + b * LAT + j) = make_float2(hn2[0], hn2[1]);
    // A-pack position for (m=b, k=j): KT = LAT/32 = 32
    int mt = b >> 4, kt = j >> 5;
    int g16  = (j >> 4) & 1;
    int half = (j >> 3) & 1;
    int w    = (j >> 1) & 3;
    int lane = half * 16 + (b & 15);
    int v    = g16 * 4 + w;
    hpack[(mt * 32 + kt) * 256 + lane * 8 + v] = pack2(hn2[0], hn2[1]);
}

// ---------------- reward head 2: per-row dot(rhid, w2) + b2 ----------------
__global__ __launch_bounds__(128) void rew2_k(const float* __restrict__ rh,
                                              const float* __restrict__ w2,
                                              const float* __restrict__ b2,
                                              float* __restrict__ out, int t) {
    int lane = threadIdx.x & 31;
    int row  = blockIdx.x * 4 + (threadIdx.x >> 5);
    float s = 0.f;
    for (int k = lane; k < RHID; k += 32) s += rh[row * RHID + k] * w2[k];
#pragma unroll
    for (int m = 16; m >= 1; m >>= 1) s += __shfl_xor(s, m, 32);
    if (lane == 0) out[row * TT + t] = s + b2[0];
}

// ---------------- launch ----------------
static inline size_t alignup(size_t x) { return (x + 255) & ~size_t(255); }

extern "C" void kernel_launch(void* const* d_in, const int* in_sizes, int n_in,
                              void* d_out, int out_size, void* d_ws, size_t ws_size,
                              hipStream_t stream) {
    const float* init_obs = (const float*)d_in[0];
    const float* actions  = (const float*)d_in[1];
    const float* enc_w1 = (const float*)d_in[2];  const float* enc_b1 = (const float*)d_in[3];
    const float* enc_w2 = (const float*)d_in[4];  const float* enc_b2 = (const float*)d_in[5];
    const float* gru_wih = (const float*)d_in[6]; const float* gru_bih = (const float*)d_in[7];
    const float* gru_whh = (const float*)d_in[8]; const float* gru_bhh = (const float*)d_in[9];
    const float* dec_w1 = (const float*)d_in[10]; const float* dec_b1 = (const float*)d_in[11];
    const float* dec_w2 = (const float*)d_in[12]; const float* dec_b2 = (const float*)d_in[13];
    const float* rew_w1 = (const float*)d_in[14]; const float* rew_b1 = (const float*)d_in[15];
    const float* rew_w2 = (const float*)d_in[16]; const float* rew_b2 = (const float*)d_in[17];

    float* obs_out = (float*)d_out;                         // (B,T,OBS)
    float* rew_out = (float*)d_out + (size_t)BB * TT * OBS; // (B,T)

    // workspace layout (bytes)
    char* ws = (char*)d_ws; size_t off = 0;
    auto take = [&](size_t bytes) { char* p = ws + off; off = alignup(off + bytes); return p; };
    char*     enc_w1p = take((size_t)OBS * HID * 2);
    char*     enc_w2p = take((size_t)HID * LAT * 2);
    char*     gru_wp  = take((size_t)RIN * GN * 2);
    char*     dec_w1p = take((size_t)LAT * HID * 2);
    char*     dec_w2p = take((size_t)HID * OBS * 2);
    char*     rew_w1p = take((size_t)LAT * RHID * 2);
    float*    gbias   = (float*)take((size_t)GN * 4);
    char*     xpack   = take((size_t)BB * RIN * 2);
    float*    gates   = (float*)take((size_t)BB * GN * 4);
    float*    hbuf    = (float*)take((size_t)BB * LAT * 4);
    char*     hpack   = take((size_t)BB * LAT * 2);
    float*    hidbuf  = (float*)take((size_t)BB * HID * 4);
    char*     hidpack = take((size_t)BB * HID * 2);
    float*    rhid    = (float*)take((size_t)BB * RHID * 4);
    (void)ws_size; (void)n_in; (void)in_sizes; (void)out_size;

    // ---- one-time weight packing (bf16 fragment layouts) ----
    pack_b_k<<<OBS * HID / 512, 256, 0, stream>>>(enc_w1, (unsigned*)enc_w1p, OBS, HID);
    pack_b_k<<<HID * LAT / 512, 256, 0, stream>>>(enc_w2, (unsigned*)enc_w2p, HID, LAT);
    pack_b_gru_k<<<RIN * GN / 512, 256, 0, stream>>>(gru_wih, gru_whh, (unsigned*)gru_wp);
    pack_b_k<<<LAT * HID / 512, 256, 0, stream>>>(dec_w1, (unsigned*)dec_w1p, LAT, HID);
    pack_b_k<<<HID * OBS / 512, 256, 0, stream>>>(dec_w2, (unsigned*)dec_w2p, HID, OBS);
    pack_b_k<<<LAT * RHID / 512, 256, 0, stream>>>(rew_w1, (unsigned*)rew_w1p, LAT, RHID);
    gbias_k<<<GN / 256, 256, 0, stream>>>(gru_bih, gru_bhh, gbias);

    const dim3 blk(128);
    // ---- encoder: h0 = relu(obs@W1+b1)@W2+b2 ----
    pack_a_k<<<BB * OBS / 512, 256, 0, stream>>>(init_obs, (unsigned*)xpack, OBS);
    gemm_bf16_wmma<OBS / 32><<<dim3(HID / 128, BB / 128), blk, 0, stream>>>(
        xpack, enc_w1p, enc_b1, hidbuf, HID, 1);
    pack_a_k<<<BB * HID / 512, 256, 0, stream>>>(hidbuf, (unsigned*)hidpack, HID);
    gemm_bf16_wmma<HID / 32><<<dim3(LAT / 128, BB / 128), blk, 0, stream>>>(
        hidpack, enc_w2p, enc_b2, hbuf, LAT, 0);

    // ---- sequential rollout ----
    for (int t = 0; t < TT; ++t) {
        pack_ax_k<<<BB * RIN / 512, 256, 0, stream>>>(hbuf, actions, t, (unsigned*)xpack);
        gemm_bf16_wmma<RIN / 32><<<dim3(GN / 128, BB / 128), blk, 0, stream>>>(
            xpack, gru_wp, gbias, gates, GN, 0);
        gru_gate_k<<<BB * LAT / 512, 256, 0, stream>>>(gates, hbuf, (unsigned*)hpack);
        // decoder
        gemm_bf16_wmma<LAT / 32><<<dim3(HID / 128, BB / 128), blk, 0, stream>>>(
            hpack, dec_w1p, dec_b1, hidbuf, HID, 1);
        pack_a_k<<<BB * HID / 512, 256, 0, stream>>>(hidbuf, (unsigned*)hidpack, HID);
        gemm_bf16_wmma<HID / 32><<<dim3(OBS / 128, BB / 128), blk, 0, stream>>>(
            hidpack, dec_w2p, dec_b2, obs_out + (size_t)t * OBS, TT * OBS, 0);
        // reward head
        gemm_bf16_wmma<LAT / 32><<<dim3(RHID / 128, BB / 128), blk, 0, stream>>>(
            hpack, rew_w1p, rew_b1, rhid, RHID, 1);
        rew2_k<<<BB / 4, 128, 0, stream>>>(rhid, rew_w2, rew_b2, rew_out, t);
    }
}
